// primal_dual_straight_71330816852238
// MI455X (gfx1250) — compile-verified, matrix operand order checked by reading, not run on hardware
//
#include <hip/hip_runtime.h>
#include <hip/hip_bf16.h>

// ---------------------------------------------------------------------------
// Learned primal-dual CT recon, MI455X (gfx1250, wave32).
// Bottleneck: streaming the two 212MB system matrices 10x each (~4.25 GB ->
// ~182us at 23.3 TB/s). Matvecs use non-temporal float4 streaming loads.
// Convs use v_wmma_f32_16x16x32_f16 implicit GEMM with LDS-staged operands
// (no divergent global gathers in the inner loop).
// ---------------------------------------------------------------------------

typedef _Float16 h16 __attribute__((ext_vector_type(16)));
typedef float    f8  __attribute__((ext_vector_type(8)));
typedef float    f4  __attribute__((ext_vector_type(4)));

#define NITER 10
#define A_ANG 60
#define HH 96
#define WW 96
#define NPIX_D (A_ANG * HH)   // 5760  dual image pixels
#define NPIX_P (HH * WW)      // 9216  primal image pixels

// ---------------------------------------------------------------------------
// init: copy dual + primal state into workspace
__global__ __launch_bounds__(256) void k_init(const float* __restrict__ dual_in,
                                              const float* __restrict__ primal_in,
                                              float* __restrict__ dual_cur,
                                              float* __restrict__ p2d) {
  int i = blockIdx.x * 256 + threadIdx.x;
  if (i < 5 * NPIX_P) p2d[i] = primal_in[i];
  if (i < NPIX_D)     dual_cur[i] = dual_in[i];
}

// fortran-flatten primal channel 1: img_f[h + 96*w] = p2d[1][h][w]; zero vec
__global__ __launch_bounds__(256) void k_fwd_pack(const float* __restrict__ p2d,
                                                  float* __restrict__ img_f,
                                                  float* __restrict__ vec) {
  int n = blockIdx.x * 256 + threadIdx.x;
  if (n < NPIX_P) {
    int h = n % HH, w = n / HH;
    img_f[n] = p2d[NPIX_P + h * WW + w];
    vec[n] = 0.0f;
  }
}

// prod[row] = dot(norm[row, 0:9216], img_f)   row = a*96 + m  (rows contiguous)
// 212 MB streamed NT; img_f (36KB) stays cache-resident.
__global__ __launch_bounds__(256) void k_fwd_proj(const float* __restrict__ norm,
                                                  const float* __restrict__ img_f,
                                                  float* __restrict__ prod) {
  __shared__ float red[256];
  const f4* rp = (const f4*)(norm + (size_t)blockIdx.x * NPIX_P);
  const f4* vp = (const f4*)img_f;
  float s = 0.0f;
  for (int i = threadIdx.x; i < NPIX_P / 4; i += 256) {
    f4 a = __builtin_nontemporal_load(rp + i);
    f4 v = vp[i];
    s += a.x * v.x + a.y * v.y + a.z * v.z + a.w * v.w;
  }
  red[threadIdx.x] = s;
  __syncthreads();
  for (int off = 128; off > 0; off >>= 1) {
    if (threadIdx.x < off) red[threadIdx.x] += red[threadIdx.x + off];
    __syncthreads();
  }
  if (threadIdx.x == 0) prod[blockIdx.x] = red[0];
}

// concat [dual, evalop1(=prod), g] -> x_dual [3, 60, 96]
__global__ __launch_bounds__(256) void k_concat_dual(const float* __restrict__ dual,
                                                     const float* __restrict__ prod,
                                                     const float* __restrict__ g,
                                                     float* __restrict__ x) {
  int i = blockIdx.x * 256 + threadIdx.x;
  if (i < NPIX_D) {
    x[i] = dual[i];
    x[NPIX_D + i] = prod[i];
    x[2 * NPIX_D + i] = g[i];
  }
}

// vec[v] += sum_{a in chunk} sum_n angles[a, v, n] * dual[a, n]
// grid (pixel-chunks, angle-chunks) for wave parallelism; per-lane rows are
// contiguous 384B (full cacheline use); dual slice cached in LDS.
#define ACHUNK 10
__global__ __launch_bounds__(256) void k_adjoint(const float* __restrict__ angles,
                                                 const float* __restrict__ dual,
                                                 float* __restrict__ vec) {
  __shared__ float dld[ACHUNK * HH];
  int a0 = blockIdx.y * ACHUNK;
  for (int i = threadIdx.x; i < ACHUNK * HH; i += 256) dld[i] = dual[a0 * HH + i];
  __syncthreads();
  int v = blockIdx.x * 256 + threadIdx.x;
  float s = 0.0f;
  for (int ai = 0; ai < ACHUNK; ++ai) {
    const f4* rp = (const f4*)(angles + ((size_t)(a0 + ai) * NPIX_P + v) * HH);
    const f4* dp = (const f4*)(dld + ai * HH);
#pragma unroll
    for (int i = 0; i < HH / 4; ++i) {
      f4 x = __builtin_nontemporal_load(rp + i);
      f4 d = dp[i];
      s += x.x * d.x + x.y * d.y + x.z * d.z + x.w * d.w;
    }
  }
  atomicAdd(&vec[v], s);
}

// xp = concat [p2d (5ch), evalop2]  with evalop2[h,w] = vec[w*96 + h]
__global__ __launch_bounds__(256) void k_pack_primal(const float* __restrict__ p2d,
                                                     const float* __restrict__ vec,
                                                     float* __restrict__ xp) {
  int i = blockIdx.x * 256 + threadIdx.x;
  if (i < 6 * NPIX_P) {
    int ch = i / NPIX_P, p = i % NPIX_P;
    if (ch < 5) xp[i] = p2d[i];
    else {
      int h = p / WW, w = p % WW;
      xp[i] = vec[w * HH + h];
    }
  }
}

__global__ __launch_bounds__(256) void k_out(const float* __restrict__ p2d,
                                             float* __restrict__ out) {
  int i = blockIdx.x * 256 + threadIdx.x;
  if (i < NPIX_P) out[i] = p2d[i];
}

// ---------------------------------------------------------------------------
// 3x3 same-pad conv as implicit GEMM on v_wmma_f32_16x16x32_f16.
// Block = 192 threads (6 full waves) = one image row (96 px, 6 M-tiles of 16).
// Input halo rows and weights are staged into LDS as f16 with coalesced
// global loads; fragment building is pure ds_load with compile-time indices
// (fully unrolled K loop), so no divergent global gathers / serialized waits.
// Fragment layouts per ISA 7.12.2 (wave32):
//   A 16x32 f16 : lane%16 = M row; VGPR j holds K pair
//                 k0 = (j<4 ? 2j : 16+2(j-4)) + 8*(lane/16)
//   B 32x16 f16 : lane%16 = N col; same j->K map (mirrored half-wave split)
//   C/D 16x16 f32: lane%16 = N; VGPR j -> M = j + 8*(lane/16)
template <int CIN, int COUT, bool PRELU, bool RESIDUAL>
__global__ void __launch_bounds__(192)
k_conv3x3_wmma(const float* __restrict__ in,   // [CIN, IH, 96]
               const float* __restrict__ wgt,  // [COUT, CIN, 3, 3]
               const float* __restrict__ bias, // [COUT]
               const float* __restrict__ alpha,// [1] prelu slope
               const float* __restrict__ res,  // [COUT, IH, 96] residual
               float* __restrict__ out,        // [COUT, IH, 96]
               int IH) {
  constexpr int K  = CIN * 9;
  constexpr int KT = (K + 31) / 32;
  constexpr int NT = (COUT + 15) / 16;   // N (out-channel) tiles
  constexpr int LROW = 100;              // padded LDS row stride (halo 98 used)
  __shared__ _Float16 sIn[CIN * 3 * LROW];
  __shared__ _Float16 sW[NT * 16 * K];

  const int tid = threadIdx.x;
  const int y = blockIdx.x;
  const int NPIX = IH * WW;

  // zero the halo pad columns
  for (int idx = tid; idx < CIN * 3; idx += 192) {
    sIn[idx * LROW + 0]  = (_Float16)0.0f;
    sIn[idx * LROW + 97] = (_Float16)0.0f;
  }
  // stage input rows y-1..y+1 (zero rows outside image), coalesced
  for (int idx = tid; idx < CIN * 3 * WW; idx += 192) {
    int ci  = idx / (3 * WW);
    int rem = idx - ci * 3 * WW;
    int ky  = rem / WW;
    int xx  = rem - ky * WW;
    int yy  = y + ky - 1;
    float v = (yy >= 0 && yy < IH) ? in[ci * NPIX + yy * WW + xx] : 0.0f;
    sIn[(ci * 3 + ky) * LROW + xx + 1] = (_Float16)v;
  }
  // stage weights [c][k] as f16, channel-padded to NT*16
  for (int idx = tid; idx < NT * 16 * K; idx += 192) {
    int c = idx / K;
    int k = idx - c * K;
    sW[idx] = (c < COUT) ? (_Float16)wgt[c * K + k] : (_Float16)0.0f;
  }
  __syncthreads();

  const int wave = tid >> 5;
  const int lane = tid & 31;
  const int hlf  = lane >> 4;
  const int l16  = lane & 15;
  const int x    = wave * 16 + l16;   // A-row pixel column in [0,96)

  f8 zero = {};
  f8 acc[NT];
#pragma unroll
  for (int t = 0; t < NT; ++t) acc[t] = zero;

#pragma unroll
  for (int kt = 0; kt < KT; ++kt) {
    h16 afrag;
    h16 bfrag[NT];
#pragma unroll
    for (int j = 0; j < 8; ++j) {
      int k0 = kt * 32 + ((j < 4) ? (2 * j) : (16 + 2 * (j - 4))) + 8 * hlf;
#pragma unroll
      for (int e = 0; e < 2; ++e) {
        int k = k0 + e;                 // compile-time after full unroll
        _Float16 av = (_Float16)0.0f;
        if (k < K) {
          int ci = k / 9, r = k % 9;
          int ky = r / 3, kx = r % 3;
          av = sIn[(ci * 3 + ky) * LROW + x + kx];
        }
        afrag[2 * j + e] = av;
#pragma unroll
        for (int t = 0; t < NT; ++t)
          bfrag[t][2 * j + e] =
              (k < K) ? sW[(t * 16 + l16) * K + k] : (_Float16)0.0f;
      }
    }
#pragma unroll
    for (int t = 0; t < NT; ++t)
      acc[t] = __builtin_amdgcn_wmma_f32_16x16x32_f16(false, afrag, false,
                                                      bfrag[t], (short)0,
                                                      acc[t], false, false);
  }

  // epilogue: lane's channel c = t*16 + lane%16, pixels wave*16 + j + 8*(lane/16)
#pragma unroll
  for (int t = 0; t < NT; ++t) {
    int c = t * 16 + l16;
    if (c < COUT) {
      float b = bias[c];
      float a = 0.0f;
      if constexpr (PRELU) a = alpha[0];
#pragma unroll
      for (int j = 0; j < 8; ++j) {
        int px = wave * 16 + j + 8 * hlf;
        int o = c * NPIX + y * WW + px;
        float v = acc[t][j] + b;
        if constexpr (PRELU) v = (v >= 0.0f) ? v : a * v;
        if constexpr (RESIDUAL) v += res[o];
        out[o] = v;
      }
    }
  }
}

// ---------------------------------------------------------------------------
extern "C" void kernel_launch(void* const* d_in, const int* in_sizes, int n_in,
                              void* d_out, int out_size, void* d_ws, size_t ws_size,
                              hipStream_t stream) {
  const float* dual_in   = (const float*)d_in[0];
  const float* primal_in = (const float*)d_in[1];
  const float* g         = (const float*)d_in[2];
  const float* angles    = (const float*)d_in[3];
  const float* norm      = (const float*)d_in[4];
  const float* dw1 = (const float*)d_in[5];
  const float* db1 = (const float*)d_in[6];
  const float* da1 = (const float*)d_in[7];
  const float* dw2 = (const float*)d_in[8];
  const float* db2 = (const float*)d_in[9];
  const float* da2 = (const float*)d_in[10];
  const float* dw3 = (const float*)d_in[11];
  const float* db3 = (const float*)d_in[12];
  const float* pw1 = (const float*)d_in[13];
  const float* pb1 = (const float*)d_in[14];
  const float* pa1 = (const float*)d_in[15];
  const float* pw2 = (const float*)d_in[16];
  const float* pb2 = (const float*)d_in[17];
  const float* pa2 = (const float*)d_in[18];
  const float* pw3 = (const float*)d_in[19];
  const float* pb3 = (const float*)d_in[20];

  float* ws       = (float*)d_ws;          // ~2.96 MB used, all 64B-aligned
  float* img_f    = ws;                    // 9216
  float* prod     = ws + 9216;             // 5760
  float* dual_cur = ws + 14976;            // 5760
  float* p2d      = ws + 20736;            // 46080
  float* vec      = ws + 66816;            // 9216
  float* x_dual   = ws + 76032;            // 17280
  float* xp       = ws + 93312;            // 55296
  float* buf1     = ws + 148608;           // 294912 (conv stage 1)
  float* buf2     = ws + 443520;           // 294912 (conv stage 2)

  k_init<<<180, 256, 0, stream>>>(dual_in, primal_in, dual_cur, p2d);

  for (int k = 0; k < NITER; ++k) {
    // ---- dual update ----
    k_fwd_pack<<<36, 256, 0, stream>>>(p2d, img_f, vec);
    k_fwd_proj<<<NPIX_D, 256, 0, stream>>>(norm, img_f, prod);
    k_concat_dual<<<23, 256, 0, stream>>>(dual_cur, prod, g, x_dual);
    k_conv3x3_wmma<3, 32, true, false><<<A_ANG, 192, 0, stream>>>(
        x_dual, dw1 + k * 864, db1 + k * 32, da1 + k, nullptr, buf1, A_ANG);
    k_conv3x3_wmma<32, 32, true, false><<<A_ANG, 192, 0, stream>>>(
        buf1, dw2 + k * 9216, db2 + k * 32, da2 + k, nullptr, buf2, A_ANG);
    k_conv3x3_wmma<32, 1, false, true><<<A_ANG, 192, 0, stream>>>(
        buf2, dw3 + k * 288, db3 + k, nullptr, dual_cur, dual_cur, A_ANG);

    // ---- primal update ----
    k_adjoint<<<dim3(NPIX_P / 256, A_ANG / ACHUNK), 256, 0, stream>>>(angles, dual_cur, vec);
    k_pack_primal<<<216, 256, 0, stream>>>(p2d, vec, xp);
    k_conv3x3_wmma<6, 32, true, false><<<HH, 192, 0, stream>>>(
        xp, pw1 + k * 1728, pb1 + k * 32, pa1 + k, nullptr, buf1, HH);
    k_conv3x3_wmma<32, 32, true, false><<<HH, 192, 0, stream>>>(
        buf1, pw2 + k * 9216, pb2 + k * 32, pa2 + k, nullptr, buf2, HH);
    k_conv3x3_wmma<32, 5, false, true><<<HH, 192, 0, stream>>>(
        buf2, pw3 + k * 1440, pb3 + k * 5, nullptr, p2d, p2d, HH);
  }

  k_out<<<36, 256, 0, stream>>>(p2d, (float*)d_out);
}
